// Attention_7567732376021
// MI455X (gfx1250) — matrix operand order repaired and checked by reading, not compiled, not run on hardware
//
#include <hip/hip_runtime.h>
#include <cstdint>
#include <cstddef>

// ---------------------------------------------------------------------------
// Types for CDNA5 WMMA (wave32): bf16 A/B fragments (16 elems) and f32 C/D (8)
// ---------------------------------------------------------------------------
typedef __bf16 bf16;
typedef __attribute__((ext_vector_type(4)))  bf16  v4bf;
typedef __attribute__((ext_vector_type(8)))  bf16  v8bf;
typedef __attribute__((ext_vector_type(16))) bf16  v16bf;
typedef __attribute__((ext_vector_type(8)))  float v8f;
typedef __attribute__((ext_vector_type(4)))  int   v4i;

#define DIM    1152
#define NH     16
#define HD     72
#define SEQ    2048
#define BATCH  2
#define ROWS   (BATCH * SEQ)      // 4096 tokens
#define QK_PAD 96                 // head_dim padded to 3*32 for K-dim of QK^T
#define V_PAD  80                 // head_dim padded to 5*16 for N-dim of AV
#define LOG2E  1.4426950408889634f
#define QSCALE 0.11785113019775793f   // 72^-0.5

// LDS strides padded for conflict-free 16-lane row access (units: bf16 elems)
#define KLS_STRIDE 104            // 96 + 8  -> 52 dwords, gcd(52,64)=4
#define VLS_STRIDE 72             // 64 + 8  -> 36 dwords, gcd(36,64)=4
#define PLS_STRIDE 72

#define AS1 __attribute__((address_space(1)))
#define AS3 __attribute__((address_space(3)))

// ---- async global->LDS copy (ASYNCcnt path) with safe fallback -------------
#if __has_builtin(__builtin_amdgcn_global_load_async_to_lds_b128)
#define HAVE_ASYNC_LDS 1
#define ASYNC_CP16(g, l)                                                     \
  __builtin_amdgcn_global_load_async_to_lds_b128((AS1 v4i*)(g),              \
                                                 (AS3 v4i*)(l), 0, 0)
#else
#define HAVE_ASYNC_LDS 0
#define ASYNC_CP16(g, l) (*(v8bf*)(l) = *(const v8bf*)(g))
#endif

static __device__ __forceinline__ void async_wait_all() {
#if HAVE_ASYNC_LDS
#if __has_builtin(__builtin_amdgcn_s_wait_asynccnt)
  __builtin_amdgcn_s_wait_asynccnt(0);
#else
  asm volatile("s_wait_asynccnt 0x0" ::: "memory");
#endif
#endif
}

static __device__ __forceinline__ v16bf load_frag(const bf16* p0, const bf16* p1) {
  v8bf lo = *(const v8bf*)p0;
  v8bf hi = *(const v8bf*)p1;
  return __builtin_shufflevector(lo, hi, 0,1,2,3,4,5,6,7,8,9,10,11,12,13,14,15);
}

static __device__ __forceinline__ v8f wmma_bf16(v16bf a, v16bf b, v8f c) {
  // (neg_a, A, neg_b, B, c_mod, C, reuse_a, reuse_b)
  return __builtin_amdgcn_wmma_f32_16x16x32_bf16(false, a, false, b, (short)0, c,
                                                 false, false);
}

static __device__ __forceinline__ float wsum32(float x) {
#pragma unroll
  for (int m = 1; m < 32; m <<= 1) x += __shfl_xor(x, m, 32);
  return x;
}
static __device__ __forceinline__ float rmax16(float x) {
#pragma unroll
  for (int m = 1; m < 16; m <<= 1) x = fmaxf(x, __shfl_xor(x, m, 32));
  return x;
}
static __device__ __forceinline__ float rsum16(float x) {
#pragma unroll
  for (int m = 1; m < 16; m <<= 1) x += __shfl_xor(x, m, 32);
  return x;
}

// ---------------------------------------------------------------------------
// K0: f32 -> bf16 conversion (vectorized x4)
// ---------------------------------------------------------------------------
__global__ __launch_bounds__(256) void cvt_f32_bf16(const float* __restrict__ in,
                                                    bf16* __restrict__ out, int n4) {
  int i = blockIdx.x * 256 + threadIdx.x;
  if (i < n4) {
    float4 v = ((const float4*)in)[i];
    v4bf r;
    r[0] = (bf16)v.x; r[1] = (bf16)v.y; r[2] = (bf16)v.z; r[3] = (bf16)v.w;
    ((v4bf*)out)[i] = r;
  }
}

// ---------------------------------------------------------------------------
// K1/K4: C[M,N] = A[M,K](bf16) * B[N,K]^T(bf16) + bias, f32 out.
// Block: 256 thr = 8 waves (2x4); wave tile 64x32 = 4x2 WMMA 16x16 tiles.
// ---------------------------------------------------------------------------
__global__ __launch_bounds__(256) void gemm_bf16_nt(const bf16* __restrict__ A,
                                                    const bf16* __restrict__ B,
                                                    const float* __restrict__ bias,
                                                    float* __restrict__ C,
                                                    int M, int N, int K) {
  const int lane = threadIdx.x & 31;
  const int wave = threadIdx.x >> 5;
  const int col  = lane & 15;
  const int hi16 = lane >> 4;       // which 16-lane half
  const int kaddA = hi16 * 8;       // A-frag K offset per half
  const int wr = wave >> 2;         // 0..1
  const int wc = wave & 3;          // 0..3
  const int rowBase = blockIdx.y * 128 + wr * 64;
  const int colBase = blockIdx.x * 128 + wc * 32;

  v8f acc[4][2];
#pragma unroll
  for (int i = 0; i < 4; ++i)
#pragma unroll
    for (int j = 0; j < 2; ++j)
#pragma unroll
      for (int e = 0; e < 8; ++e) acc[i][j][e] = 0.f;

  const bf16* Arow[4];
#pragma unroll
  for (int mt = 0; mt < 4; ++mt)
    Arow[mt] = A + (size_t)(rowBase + mt * 16 + col) * K;
  const bf16* Brow[2];
#pragma unroll
  for (int nt = 0; nt < 2; ++nt)
    Brow[nt] = B + (size_t)(colBase + nt * 16 + col) * K;

  for (int kk = 0; kk < K; kk += 32) {
    v16bf af[4], bfg[2];
#pragma unroll
    for (int mt = 0; mt < 4; ++mt) {
      const bf16* p = Arow[mt] + kk + hi16 * 16; // 16 contiguous K values
      af[mt] = load_frag(p, p + 8);
    }
#pragma unroll
    for (int nt = 0; nt < 2; ++nt) {
      const bf16* p = Brow[nt] + kk + hi16 * 16; // 16 contiguous K values
      bfg[nt] = load_frag(p, p + 8);
    }
    if (kk + 32 < K) {
      __builtin_prefetch(Arow[0] + kk + 32, 0, 1);
      __builtin_prefetch(Brow[0] + kk + 32, 0, 1);
    }
#pragma unroll
    for (int mt = 0; mt < 4; ++mt)
#pragma unroll
      for (int nt = 0; nt < 2; ++nt)
        acc[mt][nt] = wmma_bf16(af[mt], bfg[nt], acc[mt][nt]);
  }

#pragma unroll
  for (int mt = 0; mt < 4; ++mt)
#pragma unroll
    for (int nt = 0; nt < 2; ++nt) {
      const int c = colBase + nt * 16 + col;
      const float bv = bias[c];
#pragma unroll
      for (int r = 0; r < 8; ++r) {
        const int row = rowBase + mt * 16 + r + 8 * hi16;
        C[(size_t)row * N + c] = acc[mt][nt][r] + bv;
      }
    }
}

// ---------------------------------------------------------------------------
// K2: split qkv, LayerNorm q/k over D=72, pad to 96; fold SCALE*log2e into q.
// Transpose v into [bh, 80(pad), SEQ] bf16 so AV B-operand is K-contiguous.
// One wave per (b,h,n) row.
// ---------------------------------------------------------------------------
__global__ __launch_bounds__(256) void qkv_ln_split(const float* __restrict__ qkvf,
                                                    const float* __restrict__ qg,
                                                    const float* __restrict__ qbt,
                                                    const float* __restrict__ kg,
                                                    const float* __restrict__ kbt,
                                                    bf16* __restrict__ qbuf,
                                                    bf16* __restrict__ kbuf,
                                                    bf16* __restrict__ vtb) {
  const int lane = threadIdx.x & 31;
  const int warp = threadIdx.x >> 5;
  const int rid  = blockIdx.x * 8 + warp;   // 0 .. B*H*SEQ-1
  const int bh = rid >> 11;                 // b*16 + h
  const int n  = rid & 2047;
  const int b  = bh >> 4;
  const int h  = bh & 15;
  const float* base = qkvf + (size_t)(b * SEQ + n) * (3 * DIM) + h * HD;

  const int d0 = lane, d1 = lane + 32, d2 = lane + 64;
  const bool has2 = (lane < 8);

  // ---- Q: layernorm, gamma/beta, * SCALE*log2e ----
  {
    const float* p = base;
    float x0 = p[d0], x1 = p[d1], x2 = has2 ? p[d2] : 0.f;
    float mu = wsum32(x0 + x1 + x2) * (1.f / 72.f);
    float e0 = x0 - mu, e1 = x1 - mu, e2 = has2 ? x2 - mu : 0.f;
    float var = wsum32(e0 * e0 + e1 * e1 + e2 * e2) * (1.f / 72.f);
    float rstd = rsqrtf(var + 1e-5f);
    const float sc = QSCALE * LOG2E;
    bf16* out = qbuf + (size_t)(bh * SEQ + n) * QK_PAD;
    out[d0] = (bf16)((e0 * rstd * qg[d0] + qbt[d0]) * sc);
    out[d1] = (bf16)((e1 * rstd * qg[d1] + qbt[d1]) * sc);
    if (has2) out[d2] = (bf16)((e2 * rstd * qg[d2] + qbt[d2]) * sc);
    if (lane < 24) out[72 + lane] = (bf16)0.f;
  }
  // ---- K: layernorm, gamma/beta ----
  {
    const float* p = base + DIM;
    float x0 = p[d0], x1 = p[d1], x2 = has2 ? p[d2] : 0.f;
    float mu = wsum32(x0 + x1 + x2) * (1.f / 72.f);
    float e0 = x0 - mu, e1 = x1 - mu, e2 = has2 ? x2 - mu : 0.f;
    float var = wsum32(e0 * e0 + e1 * e1 + e2 * e2) * (1.f / 72.f);
    float rstd = rsqrtf(var + 1e-5f);
    bf16* out = kbuf + (size_t)(bh * SEQ + n) * QK_PAD;
    out[d0] = (bf16)(e0 * rstd * kg[d0] + kbt[d0]);
    out[d1] = (bf16)(e1 * rstd * kg[d1] + kbt[d1]);
    if (has2) out[d2] = (bf16)(e2 * rstd * kg[d2] + kbt[d2]);
    if (lane < 24) out[72 + lane] = (bf16)0.f;
  }
  // ---- V: transpose to [bh, d, n] with d padded to 80 ----
  {
    const float* p = base + 2 * DIM;
    bf16* vt = vtb + (size_t)bh * V_PAD * SEQ;
    vt[(size_t)d0 * SEQ + n] = (bf16)p[d0];
    vt[(size_t)d1 * SEQ + n] = (bf16)p[d1];
    if (has2) vt[(size_t)d2 * SEQ + n] = (bf16)p[d2];
    if (lane < 8) vt[(size_t)(72 + lane) * SEQ + n] = (bf16)0.f;
  }
}

// ---------------------------------------------------------------------------
// K3: flash attention. Block = 4 waves; each wave owns 16 q-rows of one (b,h).
// K/V tiles for each 64-key block are staged into LDS once per block via the
// async global->LDS path (ASYNCcnt), shared by all 4 waves. Per key block:
// S (16x64) via 12 WMMA, online softmax in C-layout, P -> LDS -> A-layout
// fragments, O += P*V^T via 10 WMMA (N padded to 80).
// ---------------------------------------------------------------------------
__global__ __launch_bounds__(128) void fa_kernel(const bf16* __restrict__ qbuf,
                                                 const bf16* __restrict__ kbuf,
                                                 const bf16* __restrict__ vtb,
                                                 bf16* __restrict__ obuf) {
  __shared__ bf16 kls[64 * KLS_STRIDE];        // 64 keys x 96 (+pad)
  __shared__ bf16 vls[V_PAD * VLS_STRIDE];     // 80 d    x 64 (+pad)
  __shared__ bf16 pls[4][16 * PLS_STRIDE];     // per-wave P staging

  const int tid  = threadIdx.x;
  const int lane = tid & 31;
  const int wave = tid >> 5;
  const int col  = lane & 15;
  const int hi16 = lane >> 4;
  const int kaddA = hi16 * 8;

  const int bh   = blockIdx.x >> 5;            // SEQ/64 = 32 q-blocks per bh
  const int qblk = blockIdx.x & 31;
  const int q0   = qblk * 64 + wave * 16;
  const int b = bh >> 4, h = bh & 15;

  // Q A-fragments (K = 96 -> 3 chunks of 32), loaded once from global.
  const bf16* qrowp = qbuf + ((size_t)bh * SEQ + q0) * QK_PAD + (size_t)col * QK_PAD;
  v16bf qa[3];
#pragma unroll
  for (int kk = 0; kk < 3; ++kk) {
    const bf16* p = qrowp + kk * 32 + hi16 * 16;
    qa[kk] = load_frag(p, p + 8);
  }

  float m[8], l[8];
  v8f o[5];
#pragma unroll
  for (int r = 0; r < 8; ++r) { m[r] = -3.0e38f; l[r] = 0.f; }
#pragma unroll
  for (int ot = 0; ot < 5; ++ot)
#pragma unroll
    for (int e = 0; e < 8; ++e) o[ot][e] = 0.f;

  const bf16* kbh = kbuf + (size_t)bh * SEQ * QK_PAD;
  const bf16* vbh = vtb + (size_t)bh * V_PAD * SEQ;
  bf16* pb = &pls[wave][0];

  for (int j = 0; j < SEQ; j += 64) {
    __syncthreads();   // previous tile fully consumed by all waves

    // ---- stage K tile: 64 rows x 96 elems = 768 x 16B chunks ----
#pragma unroll
    for (int c = 0; c < 6; ++c) {
      const int ch  = tid + c * 128;           // 0..767
      const int row = ch / 12;
      const int e   = (ch % 12) * 8;
      ASYNC_CP16(kbh + (size_t)(j + row) * QK_PAD + e, &kls[row * KLS_STRIDE + e]);
    }
    // ---- stage V^T tile: 80 rows x 64 elems = 640 x 16B chunks ----
#pragma unroll
    for (int c = 0; c < 5; ++c) {
      const int ch = tid + c * 128;            // 0..639
      const int d  = ch >> 3;
      const int e  = (ch & 7) * 8;
      ASYNC_CP16(vbh + (size_t)d * SEQ + j + e, &vls[d * VLS_STRIDE + e]);
    }
    async_wait_all();
    __syncthreads();

    // ---- S = Q * K^T : 4 tiles of 16 keys, K-dim 96, B-operand from LDS ----
    v8f s[4];
#pragma unroll
    for (int t = 0; t < 4; ++t) {
#pragma unroll
      for (int e = 0; e < 8; ++e) s[t][e] = 0.f;
      const int krow = (t * 16 + col) * KLS_STRIDE;
#pragma unroll
      for (int kk = 0; kk < 3; ++kk) {
        const bf16* p = &kls[krow + kk * 32 + hi16 * 16];
        v16bf bfrag = load_frag(p, p + 8);
        s[t] = wmma_bf16(qa[kk], bfrag, s[t]);
      }
    }
    // ---- online softmax (values already in log2 domain) ----
    float mn[8], alpha[8], psum[8];
#pragma unroll
    for (int r = 0; r < 8; ++r) {
      float v = fmaxf(fmaxf(s[0][r], s[1][r]), fmaxf(s[2][r], s[3][r]));
      mn[r] = fmaxf(m[r], rmax16(v));
      alpha[r] = exp2f(m[r] - mn[r]);
      m[r] = mn[r];
      psum[r] = 0.f;
    }
#pragma unroll
    for (int t = 0; t < 4; ++t)
#pragma unroll
      for (int r = 0; r < 8; ++r) {
        float pv = exp2f(s[t][r] - m[r]);
        s[t][r] = pv;
        psum[r] += pv;
      }
#pragma unroll
    for (int r = 0; r < 8; ++r)
      l[r] = l[r] * alpha[r] + rsum16(psum[r]);
#pragma unroll
    for (int ot = 0; ot < 5; ++ot)
#pragma unroll
      for (int r = 0; r < 8; ++r) o[ot][r] *= alpha[r];

    // ---- P: C-layout regs -> LDS (wave-private) -> A-layout fragments ----
#pragma unroll
    for (int t = 0; t < 4; ++t)
#pragma unroll
      for (int r = 0; r < 8; ++r)
        pb[(r + 8 * hi16) * PLS_STRIDE + t * 16 + col] = (bf16)s[t][r];
    // same-wave DS ops are in-order; compiler tracks the memory dependence
    v16bf pa[2];
#pragma unroll
    for (int kc = 0; kc < 2; ++kc) {
      const bf16* p = pb + col * PLS_STRIDE + kc * 32 + hi16 * 16;
      pa[kc] = load_frag(p, p + 8);
    }
    // ---- O += P * V^T (V^T tile in LDS, K-contiguous rows) ----
#pragma unroll
    for (int ot = 0; ot < 5; ++ot) {
      const int vrow = (ot * 16 + col) * VLS_STRIDE;
#pragma unroll
      for (int kc = 0; kc < 2; ++kc) {
        const bf16* p = &vls[vrow + kc * 32 + hi16 * 16];
        v16bf vfrag = load_frag(p, p + 8);
        o[ot] = wmma_bf16(pa[kc], vfrag, o[ot]);
      }
    }
  }

  // ---- normalize and write [B, n, h*72 + d] as bf16 for the proj GEMM ----
  float inv[8];
#pragma unroll
  for (int r = 0; r < 8; ++r) inv[r] = 1.f / l[r];
#pragma unroll
  for (int ot = 0; ot < 5; ++ot) {
    const int d = ot * 16 + col;
    if (d < HD) {
#pragma unroll
      for (int r = 0; r < 8; ++r) {
        const int qrow = q0 + r + 8 * hi16;
        obuf[((size_t)b * SEQ + qrow) * DIM + h * HD + d] = (bf16)(o[ot][r] * inv[r]);
      }
    }
  }
}

// ---------------------------------------------------------------------------
// Launch
// ---------------------------------------------------------------------------
extern "C" void kernel_launch(void* const* d_in, const int* in_sizes, int n_in,
                              void* d_out, int out_size, void* d_ws, size_t ws_size,
                              hipStream_t stream) {
  const float* x      = (const float*)d_in[0];
  const float* w_qkv  = (const float*)d_in[1];
  const float* b_qkv  = (const float*)d_in[2];
  const float* q_g    = (const float*)d_in[3];
  const float* q_b    = (const float*)d_in[4];
  const float* k_g    = (const float*)d_in[5];
  const float* k_b    = (const float*)d_in[6];
  const float* w_proj = (const float*)d_in[7];
  const float* b_proj = (const float*)d_in[8];
  float* out = (float*)d_out;

  char* ws = (char*)d_ws;
  size_t off = 0;
  auto alloc = [&](size_t bytes) -> void* {
    void* p = ws + off;
    off = (off + bytes + 255) & ~(size_t)255;
    return p;
  };
  bf16*  xb   = (bf16*)alloc((size_t)ROWS * DIM * 2);
  bf16*  wqb  = (bf16*)alloc((size_t)3 * DIM * DIM * 2);
  bf16*  wpb  = (bf16*)alloc((size_t)DIM * DIM * 2);
  float* qkvf = (float*)alloc((size_t)ROWS * 3 * DIM * 4);
  bf16*  qbuf = (bf16*)alloc((size_t)BATCH * NH * SEQ * QK_PAD * 2);
  bf16*  kbuf = (bf16*)alloc((size_t)BATCH * NH * SEQ * QK_PAD * 2);
  bf16*  vtb  = (bf16*)alloc((size_t)BATCH * NH * V_PAD * SEQ * 2);
  bf16*  obuf = (bf16*)alloc((size_t)ROWS * DIM * 2);

  {
    int n4 = ROWS * DIM / 4;
    cvt_f32_bf16<<<(n4 + 255) / 256, 256, 0, stream>>>(x, xb, n4);
  }
  {
    int n4 = 3 * DIM * DIM / 4;
    cvt_f32_bf16<<<(n4 + 255) / 256, 256, 0, stream>>>(w_qkv, wqb, n4);
  }
  {
    int n4 = DIM * DIM / 4;
    cvt_f32_bf16<<<(n4 + 255) / 256, 256, 0, stream>>>(w_proj, wpb, n4);
  }

  gemm_bf16_nt<<<dim3(3 * DIM / 128, ROWS / 128), 256, 0, stream>>>(
      xb, wqb, b_qkv, qkvf, ROWS, 3 * DIM, DIM);

  qkv_ln_split<<<BATCH * NH * SEQ / 8, 256, 0, stream>>>(
      qkvf, q_g, q_b, k_g, k_b, qbuf, kbuf, vtb);

  fa_kernel<<<BATCH * NH * (SEQ / 64), 128, 0, stream>>>(qbuf, kbuf, vtb, obuf);

  gemm_bf16_nt<<<dim3(DIM / 128, ROWS / 128), 256, 0, stream>>>(
      obuf, wpb, b_proj, out, ROWS, DIM, DIM);

  (void)in_sizes; (void)n_in; (void)out_size; (void)ws_size;
}
